// CrossTemporalAttentionModule_7395933683826
// MI455X (gfx1250) — matrix operand order repaired and checked by reading, not compiled, not run on hardware
//
#include <hip/hip_runtime.h>
#include <hip/hip_bf16.h>

// Problem constants (match reference)
#define BB   8
#define CC   128
#define HH   64
#define WWW  64
#define NN   (HH * WWW)   // 4096
#define MIDC 32

typedef __bf16          v16bf   __attribute__((ext_vector_type(16)));
typedef float           v8f     __attribute__((ext_vector_type(8)));
typedef float           float4v __attribute__((ext_vector_type(4)));
typedef unsigned short  v16us   __attribute__((ext_vector_type(16)));
typedef unsigned int    uint4v  __attribute__((ext_vector_type(4)));
typedef unsigned int    uint2v  __attribute__((ext_vector_type(2)));
typedef int             int8v   __attribute__((ext_vector_type(8)));
typedef int             int4v   __attribute__((ext_vector_type(4)));

#if __has_builtin(__builtin_amdgcn_tensor_load_to_lds)
#define USE_TDM 1
#else
#define USE_TDM 0
#endif

union BFrag {
    v16bf  bf;
    v16us  us;
    uint4v q[2];
};

__device__ __forceinline__ unsigned short f2bf(float f) {
    union { float f; unsigned u; } v; v.f = f;
    unsigned r = v.u + 0x7FFFu + ((v.u >> 16) & 1u);   // RNE
    return (unsigned short)(r >> 16);
}

// ---- DPP row_xmask cross-lane ops (stay within 16-lane rows; pure VALU) ----
__device__ __forceinline__ float dpp_xor1(float x) {
    return __builtin_bit_cast(float, __builtin_amdgcn_update_dpp(
        __builtin_bit_cast(int, x), __builtin_bit_cast(int, x), 0x161, 0xF, 0xF, false));
}
__device__ __forceinline__ float dpp_xor2(float x) {
    return __builtin_bit_cast(float, __builtin_amdgcn_update_dpp(
        __builtin_bit_cast(int, x), __builtin_bit_cast(int, x), 0x162, 0xF, 0xF, false));
}
__device__ __forceinline__ float dpp_xor4(float x) {
    return __builtin_bit_cast(float, __builtin_amdgcn_update_dpp(
        __builtin_bit_cast(int, x), __builtin_bit_cast(int, x), 0x164, 0xF, 0xF, false));
}
__device__ __forceinline__ float dpp_xor8(float x) {
    return __builtin_bit_cast(float, __builtin_amdgcn_update_dpp(
        __builtin_bit_cast(int, x), __builtin_bit_cast(int, x), 0x168, 0xF, 0xF, false));
}
__device__ __forceinline__ float row16_max(float x) {
    x = fmaxf(x, dpp_xor1(x));
    x = fmaxf(x, dpp_xor2(x));
    x = fmaxf(x, dpp_xor4(x));
    x = fmaxf(x, dpp_xor8(x));
    return x;
}
__device__ __forceinline__ float row16_sum(float x) {
    x += dpp_xor1(x);
    x += dpp_xor2(x);
    x += dpp_xor4(x);
    x += dpp_xor8(x);
    return x;
}

// ---------------------------------------------------------------------------
// Projection Q,K : per-pixel GEMV (C=128 -> MID=32 each), output bf16
// Qb/Kb layout: [b][n][mid]. Softmax scale folded into Q.
// ---------------------------------------------------------------------------
__global__ void proj_qk_kernel(const float* __restrict__ f0,
                               const float* __restrict__ f1,
                               const float* __restrict__ Wq, const float* __restrict__ bq,
                               const float* __restrict__ Wk, const float* __restrict__ bk,
                               unsigned short* __restrict__ Qb,
                               unsigned short* __restrict__ Kb) {
    const int id = blockIdx.x * blockDim.x + threadIdx.x;
    const int b  = id >> 12;
    const int n  = id & (NN - 1);
    float qa[MIDC], ka[MIDC];
#pragma unroll
    for (int m = 0; m < MIDC; ++m) { qa[m] = bq[m]; ka[m] = bk[m]; }
    const float* x0 = f0 + (size_t)b * CC * NN + n;
    const float* x1 = f1 + (size_t)b * CC * NN + n;
    for (int c = 0; c < CC; ++c) {
        const float a0 = x0[(size_t)c * NN];
        const float a1 = x1[(size_t)c * NN];
#pragma unroll
        for (int m = 0; m < MIDC; ++m) {
            qa[m] = fmaf(Wq[m * CC + c], a0, qa[m]);
            ka[m] = fmaf(Wk[m * CC + c], a1, ka[m]);
        }
    }
    const float scale = 0.088388347648318447f;   // 1/sqrt(128)
    unsigned short* qd = Qb + ((size_t)b * NN + n) * MIDC;
    unsigned short* kd = Kb + ((size_t)b * NN + n) * MIDC;
#pragma unroll
    for (int m = 0; m < MIDC; ++m) {
        qd[m] = f2bf(qa[m] * scale);
        kd[m] = f2bf(ka[m]);
    }
}

// ---------------------------------------------------------------------------
// Projection V : per-pixel GEMV (128 -> 128), 4 channel groups.
// Vb layout: [b][c][n]  (channel-major, so flash V^T tiles are row copies)
// ---------------------------------------------------------------------------
__global__ void proj_v_kernel(const float* __restrict__ f1,
                              const float* __restrict__ Wv, const float* __restrict__ bv,
                              unsigned short* __restrict__ Vb) {
    const int id = blockIdx.x * blockDim.x + threadIdx.x;
    const int b  = id >> 12;
    const int n  = id & (NN - 1);
    const int cg = blockIdx.y * 32;
    float acc[32];
#pragma unroll
    for (int m = 0; m < 32; ++m) acc[m] = bv[cg + m];
    const float* x1 = f1 + (size_t)b * CC * NN + n;
    for (int c = 0; c < CC; ++c) {
        const float a1 = x1[(size_t)c * NN];
#pragma unroll
        for (int m = 0; m < 32; ++m)
            acc[m] = fmaf(Wv[(cg + m) * CC + c], a1, acc[m]);
    }
    unsigned short* vd = Vb + ((size_t)b * CC + cg) * NN + n;
#pragma unroll
    for (int m = 0; m < 32; ++m) vd[(size_t)m * NN] = f2bf(acc[m]);
}

// ---------------------------------------------------------------------------
// Spatial mean of |f1 - f0| per (b,c)
// ---------------------------------------------------------------------------
__global__ void pool_kernel(const float* __restrict__ f0, const float* __restrict__ f1,
                            float* __restrict__ pooled) {
    const int bc = blockIdx.x;
    const float* p0 = f0 + (size_t)bc * NN;
    const float* p1 = f1 + (size_t)bc * NN;
    float s = 0.f;
    for (int i = threadIdx.x; i < NN; i += blockDim.x)
        s += fabsf(p1[i] - p0[i]);
    __shared__ float red[256];
    red[threadIdx.x] = s;
    __syncthreads();
    for (int off = 128; off > 0; off >>= 1) {
        if (threadIdx.x < off) red[threadIdx.x] += red[threadIdx.x + off];
        __syncthreads();
    }
    if (threadIdx.x == 0) pooled[bc] = red[0] * (1.0f / (float)NN);
}

// ---------------------------------------------------------------------------
// SE gate: dw[b,c] = sigmoid(Wd2 @ relu(Wd1 @ pooled + bd1) + bd2)
// ---------------------------------------------------------------------------
__global__ void gate_kernel(const float* __restrict__ pooled,
                            const float* __restrict__ Wd1, const float* __restrict__ bd1,
                            const float* __restrict__ Wd2, const float* __restrict__ bd2,
                            float* __restrict__ dw) {
    const int b = blockIdx.x;
    __shared__ float hm[MIDC];
    if (threadIdx.x < MIDC) {
        float s = bd1[threadIdx.x];
        for (int c = 0; c < CC; ++c)
            s = fmaf(Wd1[threadIdx.x * CC + c], pooled[b * CC + c], s);
        hm[threadIdx.x] = fmaxf(s, 0.f);
    }
    __syncthreads();
    const int c = threadIdx.x;
    float s = bd2[c];
#pragma unroll
    for (int m = 0; m < MIDC; ++m)
        s = fmaf(Wd2[c * MIDC + m], hm[m], s);
    dw[b * CC + c] = 1.0f / (1.0f + __expf(-s));
}

// ---------------------------------------------------------------------------
// Fused flash attention + epilogue (double-buffered K/V tiles).
// ---------------------------------------------------------------------------
#define LP 40          // padded LDS row stride (ushorts): 64B data + 16B pad
#define NT (NN / 32)   // key tiles

#if USE_TDM
// Build a 2D-tile TDM descriptor and issue tensor_load_to_lds (6-arg builtin).
// Tile: tile_d1 rows x tile_d0 elements (2B each), row stride stride0 elems.
// LDS padding: 16B after every 64B -> matches LP=40 ushort rows.
__device__ __forceinline__ void tdm_load_2d(const unsigned short* gsrc,
                                            unsigned lds_off,
                                            unsigned tile_d0, unsigned tile_d1,
                                            unsigned tensor_d1, unsigned stride0) {
    const unsigned long long ga = (unsigned long long)(uintptr_t)gsrc;
    uint4v g0;
    g0[0] = 1u;                                           // count=1, user mode
    g0[1] = lds_off;                                      // lds_addr
    g0[2] = (unsigned)(ga & 0xFFFFFFFFull);               // global_addr lo
    g0[3] = (unsigned)((ga >> 32) & 0x01FFFFFFull)        // global_addr hi (57b)
            | (2u << 30);                                 // type = 2 (image)
    int8v g1;
    g1[0] = (int)((1u << 16)      // data_size = 2B
                | (1u << 20)      // pad_enable
                | (3u << 22)      // pad_interval: 16 DWORDs (64B)
                | (3u << 25));    // pad_amount: 4 DWORDs (16B)
    g1[1] = (int)(tile_d0 << 16);                         // tensor_dim0 = tile_d0
    g1[2] = (int)((tensor_d1 & 0xFFFFu) << 16);           // tensor_dim1 lo
    g1[3] = (int)(((tensor_d1 >> 16) & 0xFFFFu) | (tile_d0 << 16)); // dim1 hi | tile_dim0
    g1[4] = (int)(tile_d1 & 0xFFFFu);                     // tile_dim1 (tile_dim2 = 0)
    g1[5] = (int)stride0;                                 // tensor_dim0_stride lo32
    g1[6] = 0;
    g1[7] = 0;
    int4v z4 = {0, 0, 0, 0};
    int8v z8 = {0, 0, 0, 0, 0, 0, 0, 0};
    __builtin_amdgcn_tensor_load_to_lds(g0, g1, z4, z4, z8, 0);
}
#endif

__launch_bounds__(256, 1)
__global__ void flash_attn_kernel(const unsigned short* __restrict__ Qb,
                                  const unsigned short* __restrict__ Kb,
                                  const unsigned short* __restrict__ Vb,
                                  const float* __restrict__ f0,
                                  const float* __restrict__ f1,
                                  const float* __restrict__ dw,
                                  const float* __restrict__ gamma,
                                  float* __restrict__ out) {
    const int b     = blockIdx.y;
    const int qbase = blockIdx.x * 128;
    const int tid   = threadIdx.x;
    const int wave  = tid >> 5;
    const int lane  = tid & 31;
    const int g     = lane >> 4;
    const int ln    = lane & 15;

    __shared__ __align__(16) unsigned short Ks[2 * 32 * LP];    // [buf][key][mid]
    __shared__ __align__(16) unsigned short Vs[2 * 128 * LP];   // [buf][ch][key]
    __shared__ __align__(16) unsigned short Ps[8][16 * LP];     // per-wave P staging

    // ---- Q fragment (A-matrix 16x32 bf16), loaded once ----
    const int q0 = qbase + wave * 16 + ln;
    const unsigned short* qrow = Qb + ((size_t)b * NN + q0) * MIDC;
    BFrag Qf;
    {
        const int k0 = g ? 16 : 0;
        const int k1 = g ? 24 : 8;
        Qf.q[0] = *(const uint4v*)(qrow + k0);
        Qf.q[1] = *(const uint4v*)(qrow + k1);
    }

    v8f acc[8];
#pragma unroll
    for (int ct = 0; ct < 8; ++ct)
#pragma unroll
        for (int r = 0; r < 8; ++r) acc[ct][r] = 0.f;
    float mrow[8], lrow[8];
#pragma unroll
    for (int r = 0; r < 8; ++r) { mrow[r] = -1e30f; lrow[r] = 0.f; }

    const unsigned short* Kbase = Kb + (size_t)b * NN * MIDC;
    const unsigned short* Vbase = Vb + (size_t)b * NN * CC;   // [c][n]

#if USE_TDM
    const unsigned ksOff[2] = { (unsigned)(size_t)&Ks[0], (unsigned)(size_t)&Ks[32 * LP] };
    const unsigned vsOff[2] = { (unsigned)(size_t)&Vs[0], (unsigned)(size_t)&Vs[128 * LP] };
    if (wave == 0) {   // prologue: DMA tile 0
        tdm_load_2d(Kbase, ksOff[0], 32, 32, NN, MIDC);
        tdm_load_2d(Vbase, vsOff[0], 32, 128, CC, NN);
    }
#endif

    for (int kt = 0; kt < NT; ++kt) {
        const int kb  = kt * 32;
        const int buf = kt & 1;
        unsigned short* KsT = Ks + buf * 32 * LP;
        unsigned short* VsT = Vs + buf * 128 * LP;

#if USE_TDM
        if (wave == 0) __builtin_amdgcn_s_wait_tensorcnt(0);   // tile kt landed
        __syncthreads();                                       // visible + prev compute done
        if (wave == 0 && kt + 1 < NT) {                        // DMA tile kt+1 during compute
            const int nkb = kb + 32;
            tdm_load_2d(Kbase + (size_t)nkb * MIDC, ksOff[buf ^ 1], 32, 32, NN, MIDC);
            tdm_load_2d(Vbase + nkb,                vsOff[buf ^ 1], 32, 128, CC, NN);
        }
#else
        __syncthreads();
        {   // K tile: 32 rows x 64B = 128 16B-chunks; threads 0..127
            if (tid < 128) {
                const int row = tid >> 2, qd = tid & 3;
                *(uint4v*)&KsT[row * LP + qd * 8] =
                    *(const uint4v*)(Kbase + (size_t)(kb + row) * MIDC + qd * 8);
            }
            // V tile: 128 rows x 64B = 512 16B-chunks; 2 per thread
#pragma unroll
            for (int it = 0; it < 2; ++it) {
                const int c2 = tid + it * 256;
                const int row = c2 >> 2, qd = c2 & 3;
                *(uint4v*)&VsT[row * LP + qd * 8] =
                    *(const uint4v*)(Vbase + (size_t)row * NN + kb + qd * 8);
            }
        }
        if (kt + 1 < NT) {
            __builtin_prefetch(Kbase + (size_t)(kb + 32 + (tid >> 3)) * MIDC, 0, 0);
            __builtin_prefetch(Vbase + (size_t)(tid >> 1) * NN + kb + 32, 0, 0);
        }
        __syncthreads();
#endif

        // ---- S = (scaled Q)^T K, two 16-key sub-tiles ----
        v8f S[2];
#pragma unroll
        for (int j = 0; j < 2; ++j) {
            BFrag Kf;   // B 32x16: lane<16: col=ln, K=0..15; lane>=16: col=ln, K=16..31
            const unsigned short* krow = KsT + (j * 16 + ln) * LP + g * 16;
            Kf.q[0] = *(const uint4v*)(krow);
            Kf.q[1] = *(const uint4v*)(krow + 8);
            v8f z;
#pragma unroll
            for (int r = 0; r < 8; ++r) z[r] = 0.f;
            S[j] = __builtin_amdgcn_wmma_f32_16x16x32_bf16(
                       false, Qf.bf, false, Kf.bf, (short)0, z, false, false);
        }

        // ---- online softmax (rows live across each 16-lane group) ----
        float alpha[8];
#pragma unroll
        for (int r = 0; r < 8; ++r) {
            const float tmax = row16_max(fmaxf(S[0][r], S[1][r]));
            const float mn   = fmaxf(mrow[r], tmax);
            alpha[r] = __expf(mrow[r] - mn);
            mrow[r]  = mn;
            S[0][r]  = __expf(S[0][r] - mn);
            S[1][r]  = __expf(S[1][r] - mn);
            lrow[r]  = lrow[r] * alpha[r] + row16_sum(S[0][r] + S[1][r]);
        }
#pragma unroll
        for (int ct = 0; ct < 8; ++ct)
#pragma unroll
            for (int r = 0; r < 8; ++r)
                acc[ct][r] *= alpha[r];

        // ---- lane-transpose P through per-wave LDS: C/D layout -> A layout ----
        unsigned short* pw = Ps[wave];
#pragma unroll
        for (int j = 0; j < 2; ++j)
#pragma unroll
            for (int r = 0; r < 8; ++r)
                pw[(r + 8 * g) * LP + (j * 16 + ln)] = f2bf(S[j][r]);
        asm volatile("s_wait_dscnt 0" ::: "memory");
        BFrag Pf;
        {
            const int k0 = g ? 16 : 0;
            const int k1 = g ? 24 : 8;
            Pf.q[0] = *(const uint4v*)(pw + ln * LP + k0);
            Pf.q[1] = *(const uint4v*)(pw + ln * LP + k1);
        }

        // ---- O[ct] += P(16x32) * V^T(32x16), 8 channel tiles ----
#pragma unroll
        for (int ct = 0; ct < 8; ++ct) {
            BFrag Vf;
            const unsigned short* vrow = VsT + (ct * 16 + ln) * LP + g * 16;
            Vf.q[0] = *(const uint4v*)(vrow);
            Vf.q[1] = *(const uint4v*)(vrow + 8);
            acc[ct] = __builtin_amdgcn_wmma_f32_16x16x32_bf16(
                          false, Pf.bf, false, Vf.bf, (short)0, acc[ct], false, false);
        }
    }

    // ---- fused epilogue: out = f0 + gamma*attended + |f1-f0|*dw ----
    const float gam = gamma[0];
    float invl[8];
#pragma unroll
    for (int r = 0; r < 8; ++r) invl[r] = 1.0f / lrow[r];
    const int qs = qbase + wave * 16 + 8 * g;   // 8 consecutive rows per lane
#pragma unroll
    for (int ct = 0; ct < 8; ++ct) {
        const int ch = ct * 16 + ln;
        const float w = dw[b * CC + ch];
        const size_t base = ((size_t)b * CC + ch) * NN + qs;
        const float4v x0a = *(const float4v*)(f0 + base);
        const float4v x0b = *(const float4v*)(f0 + base + 4);
        const float4v x1a = *(const float4v*)(f1 + base);
        const float4v x1b = *(const float4v*)(f1 + base + 4);
        float4v ra, rb;
#pragma unroll
        for (int r = 0; r < 4; ++r) {
            ra[r] = x0a[r] + gam * (acc[ct][r] * invl[r])         + fabsf(x1a[r] - x0a[r]) * w;
            rb[r] = x0b[r] + gam * (acc[ct][r + 4] * invl[r + 4]) + fabsf(x1b[r] - x0b[r]) * w;
        }
        *(float4v*)(out + base)     = ra;
        *(float4v*)(out + base + 4) = rb;
    }
}

// ---------------------------------------------------------------------------
extern "C" void kernel_launch(void* const* d_in, const int* in_sizes, int n_in,
                              void* d_out, int out_size, void* d_ws, size_t ws_size,
                              hipStream_t stream) {
    (void)in_sizes; (void)n_in; (void)out_size; (void)ws_size;
    const float* f0   = (const float*)d_in[0];
    const float* f1   = (const float*)d_in[1];
    const float* Wq   = (const float*)d_in[2];
    const float* bq   = (const float*)d_in[3];
    const float* Wk   = (const float*)d_in[4];
    const float* bk   = (const float*)d_in[5];
    const float* Wv   = (const float*)d_in[6];
    const float* bv   = (const float*)d_in[7];
    const float* Wd1  = (const float*)d_in[8];
    const float* bd1  = (const float*)d_in[9];
    const float* Wd2  = (const float*)d_in[10];
    const float* bd2  = (const float*)d_in[11];
    const float* gam  = (const float*)d_in[12];
    float* out = (float*)d_out;

    char* ws = (char*)d_ws;
    const size_t qk_bytes = (size_t)BB * NN * MIDC * sizeof(unsigned short); // 2 MB
    const size_t v_bytes  = (size_t)BB * NN * CC   * sizeof(unsigned short); // 8 MB
    unsigned short* Qb = (unsigned short*)(ws);
    unsigned short* Kb = (unsigned short*)(ws + qk_bytes);
    unsigned short* Vb = (unsigned short*)(ws + 2 * qk_bytes);
    float* pooled = (float*)(ws + 2 * qk_bytes + v_bytes);
    float* dwp    = pooled + BB * CC;

    proj_qk_kernel<<<dim3((BB * NN) / 256), dim3(256), 0, stream>>>(
        f0, f1, Wq, bq, Wk, bk, Qb, Kb);
    proj_v_kernel<<<dim3((BB * NN) / 256, 4), dim3(256), 0, stream>>>(
        f1, Wv, bv, Vb);
    pool_kernel<<<dim3(BB * CC), dim3(256), 0, stream>>>(f0, f1, pooled);
    gate_kernel<<<dim3(BB), dim3(128), 0, stream>>>(pooled, Wd1, bd1, Wd2, bd2, dwp);
    flash_attn_kernel<<<dim3(NN / 128, BB), dim3(256), 0, stream>>>(
        Qb, Kb, Vb, f0, f1, dwp, gam, out);
}
